// AntHill_90177133347089
// MI455X (gfx1250) — matrix operand order repaired and checked by reading, not compiled, not run on hardware
//
#include <hip/hip_runtime.h>
#include <math.h>

// ---------------------------------------------------------------------------
// Hyena operator for MI455X (gfx1250): blocked-Toeplitz causal convolution
// evaluated with v_wmma_f32_16x16x32_bf16. B=4, L=8192, D=1024.
// One wave owns 4 C accumulators (16 output tiles x 4 batches) so each
// Toeplitz A-build feeds 4 WMMAs. LDS k-copies are stored half-swapped so
// A operands come out of ds_load_b32 with zero fixup.
// ---------------------------------------------------------------------------

#define LSEQ   8192
#define DCH    1024
#define NBATCH 4
#define ORDER  64
#define EMBD   33
#define BANDS  16

typedef __bf16 bf16_t;
typedef __attribute__((ext_vector_type(16))) __bf16   v16bf;
typedef __attribute__((ext_vector_type(8)))  float    v8f;
typedef __attribute__((ext_vector_type(4)))  unsigned v4u;

__device__ __forceinline__ unsigned swap16(unsigned v) {
  return (v >> 16) | (v << 16);
}

// -------------------------------------------------------------------
// Kernel 1: positional features + 2-layer Sin MLP -> H [L, 64] (f32)
// -------------------------------------------------------------------
__global__ __launch_bounds__(256) void hfilt_kernel(
    const float* __restrict__ w1, const float* __restrict__ b1,
    const float* __restrict__ w2, const float* __restrict__ b2,
    float* __restrict__ H) {
  int t = blockIdx.x * 256 + threadIdx.x;
  if (t >= LSEQ) return;

  float z[EMBD];
  z[0] = (float)t / (float)(LSEQ - 1);
  float w = 2.0f * 3.14159265358979323846f * (float)t / (float)LSEQ;
  #pragma unroll
  for (int bband = 0; bband < BANDS; ++bband) {
    float f = 1e-4f + (float)bband * ((15.0f - 1e-4f) / 15.0f);
    z[1 + bband]         = __cosf(f * w);
    z[1 + BANDS + bband] = __sinf(f * w);
  }
  float h1[ORDER];
  #pragma unroll 4
  for (int o = 0; o < ORDER; ++o) {
    float acc = b1[o];
    #pragma unroll
    for (int e = 0; e < EMBD; ++e) acc += z[e] * w1[e * ORDER + o];
    h1[o] = __sinf(10.0f * acc);
  }
  float* Hr = H + (size_t)t * ORDER;
  #pragma unroll 4
  for (int o = 0; o < ORDER; ++o) {
    float acc = b2[o];
    #pragma unroll
    for (int e = 0; e < ORDER; ++e) acc += h1[e] * w2[e * ORDER + o];
    Hr[o] = __sinf(10.0f * acc);
  }
}

// -------------------------------------------------------------------
// Kernel 2: kT[d, t] = H[t,:] @ w3[:,d] + b3[d]   (bf16, [D, L])
// -------------------------------------------------------------------
__global__ __launch_bounds__(256) void kproj_kernel(
    const float* __restrict__ H, const float* __restrict__ w3,
    const float* __restrict__ b3, bf16_t* __restrict__ kT) {
  int t = blockIdx.x * 256 + threadIdx.x;
  int d = blockIdx.y;
  float acc = b3[d];
  const float* Hr = H + (size_t)t * ORDER;
  #pragma unroll
  for (int o = 0; o < ORDER; ++o) acc += Hr[o] * w3[(size_t)o * DCH + d];
  kT[(size_t)d * LSEQ + t] = (bf16_t)acc;
}

// -------------------------------------------------------------------
// Kernel 3: xT[b, d, t] = bf16(x[b, t, d] + pe[t, d])   ([B, D, L])
// -------------------------------------------------------------------
__global__ __launch_bounds__(256) void xtrans_kernel(
    const float* __restrict__ x, const float* __restrict__ pe,
    bf16_t* __restrict__ xT) {
  __shared__ float tile[32][33];
  int dT = blockIdx.x * 32;
  int tT = blockIdx.y * 32;
  int b  = blockIdx.z;
  int lx = threadIdx.x & 31;
  int ly = threadIdx.x >> 5;
  const float* xb = x + (size_t)b * LSEQ * DCH;
  #pragma unroll
  for (int r = 0; r < 32; r += 8) {
    int t = tT + ly + r;
    int d = dT + lx;
    tile[ly + r][lx] = xb[(size_t)t * DCH + d] + pe[(size_t)t * DCH + d];
  }
  __syncthreads();
  #pragma unroll
  for (int r = 0; r < 32; r += 8) {
    int d = dT + ly + r;
    int t = tT + lx;
    xT[((size_t)b * DCH + d) * LSEQ + t] = (bf16_t)tile[lx][ly + r];
  }
}

// -------------------------------------------------------------------
// Kernel 4: blocked-Toeplitz causal conv, 4 accumulators per wave.
//   Block = 1 channel d, 4 waves. Wave owns 16 same-parity output
//   tiles: Ibase + 8a + {0,2,4,6},  a = accumulator 0..3.
//   LDS holds the k row twice, HALF-SWAPPED: K0s, and K1s = K0 shifted
//   one element then half-swapped, so each descending Toeplitz pair is
//   exactly one aligned ds_load_b32 (no fixup).
//   Main loop (D2 <= Ibase): all columns valid, no masking.
//   Tail (15 iters): per-accumulator wave-uniform skip + lane masks.
// -------------------------------------------------------------------
#define K0PAD 32                    // front zero pad (negative lags)
#define K0LEN (K0PAD + LSEQ)        // 8224 elements = 4112 words
#define K1LEN (K0LEN + 2)           // 8226 elements = 4113 words

__global__ __launch_bounds__(128) void hyena_conv_wmma_kernel(
    const bf16_t* __restrict__ kT, const bf16_t* __restrict__ xT,
    bf16_t* __restrict__ yT) {
  __shared__ bf16_t shk[K0LEN + K1LEN + 2];   // K0s [0..8224), K1s [8224..16450)
  __shared__ float  shc[4][4 * 272];          // [wave][acc][16 cols x 17]

  const int bid  = blockIdx.x;
  const int d    = bid >> 3;          // channel
  const int gc   = bid & 7;           // group chunk (4 waves per block)
  const int tid  = threadIdx.x;
  const int wave = tid >> 5;
  const int lane = tid & 31;

  // --- cooperative vectorized fill (both copies stored half-swapped) ---
  const bf16_t*   krow = kT + (size_t)d * LSEQ;
  const unsigned* kw   = (const unsigned*)krow;     // 4096 words
  unsigned*       shw  = (unsigned*)shk;

  if (tid < 16) shw[tid] = 0;                       // K0 zero pad (words 0..15)
  {
    v4u*       shq = (v4u*)(shw + 16);              // byte 64: 16B aligned
    const v4u* kq  = (const v4u*)kw;
    #pragma unroll
    for (int n = 0; n < 8; ++n) {
      v4u v = kq[tid + 128 * n];
      v.x = swap16(v.x); v.y = swap16(v.y);
      v.z = swap16(v.z); v.w = swap16(v.w);
      shq[tid + 128 * n] = v;
    }
  }
  {
    unsigned* shw1 = shw + (K0LEN / 2);             // K1s word region (4113 words)
    if (tid < 16) shw1[tid] = 0;                    // K1 elements 0..31
    if (tid == 0) shw1[4112] = 0;                   // unused last word
    for (int n2 = 16 + tid; n2 < 4112; n2 += 128) {
      unsigned w0 = (n2 == 16) ? 0u : kw[n2 - 17];
      unsigned w1 = kw[n2 - 16];
      // shifted + swapped: low = lo(w1), high = hi(w0)
      shw1[n2] = (w1 & 0x0000FFFFu) | (w0 & 0xFFFF0000u);
    }
  }
  __syncthreads();

  const int g     = gc * 4 + wave;    // 0..31
  const int p     = g & 1;            // parity
  const int qq    = g >> 1;           // 0..15
  const int Ibase = p + 32 * qq;

  const int i = lane & 15;            // A row / B,C column index
  const int h = lane >> 4;            // lane half

  // per-lane LDS base: odd rows -> K0s, even rows -> K1s shifted (aligned)
  const char* shbytes = (const char*)shk;
  const int   Pbyte   = (i & 1) ? 0 : (K0LEN * 2 + 2);

  // column -> (batch, tile-in-acc)
  const int b  = i & 3;
  const int mm = i >> 2;

  const bf16_t* xrow = xT + ((size_t)b * DCH + d) * LSEQ;

  v8f cacc[4];
  #pragma unroll
  for (int a = 0; a < 4; ++a) cacc[a] = (v8f){0.f,0.f,0.f,0.f,0.f,0.f,0.f,0.f};

  const int s0 = 15 - 4 * h;          // dword index base for A loads

  // ---------------- main loop: every column of every acc is valid -----------
  #pragma unroll 2
  for (int D2 = p; D2 <= Ibase; D2 += 2) {
    // A (16x32 Toeplitz slice; A[i][j] = k[16*D2 + i - j]); words pre-swapped
    const int qbase = 16 * D2 + K0PAD + i - 1;
    const unsigned* abase =
        (const unsigned*)(shbytes + (Pbyte + 2 * (qbase - 30)));
    union { unsigned u[8]; v16bf v; } A;
    A.u[0] = abase[s0];
    A.u[1] = abase[s0 - 1];
    A.u[2] = abase[s0 - 2];
    A.u[3] = abase[s0 - 3];
    A.u[4] = abase[s0 - 8];
    A.u[5] = abase[s0 - 9];
    A.u[6] = abase[s0 - 10];
    A.u[7] = abase[s0 - 11];

    // B base for acc 0; acc a is +128 elements (+256 B)
    const int jb0 = (Ibase + 2 * mm - D2) >> 1;
    const bf16_t* xb0 = xrow + 32 * jb0 + 8 * h;
    #pragma unroll
    for (int a = 0; a < 4; ++a) {
      const v4u* xq = (const v4u*)(xb0 + 128 * a);
      union { v4u q2[2]; v16bf v; } B;
      B.q2[0] = xq[0];
      B.q2[1] = xq[2];
      cacc[a] = __builtin_amdgcn_wmma_f32_16x16x32_bf16(
          false, A.v, false, B.v, (short)0, cacc[a], false, false);
    }
  }

  // ---------------- tail: 15 masked iterations ------------------------------
  for (int D2 = Ibase + 2; D2 <= Ibase + 30; D2 += 2) {
    const int qbase = 16 * D2 + K0PAD + i - 1;
    const unsigned* abase =
        (const unsigned*)(shbytes + (Pbyte + 2 * (qbase - 30)));
    union { unsigned u[8]; v16bf v; } A;
    #pragma unroll
    for (int pr = 0; pr < 8; ++pr) {
      int idx = (pr < 4) ? (s0 - pr) : (s0 - 4 - pr);
      A.u[pr] = abase[idx];
    }
    #pragma unroll
    for (int a = 0; a < 4; ++a) {
      if (D2 <= Ibase + 8 * a + 6) {          // wave-uniform acc guard
        const int ImA = Ibase + 8 * a + 2 * mm;
        unsigned valid = (D2 <= ImA) ? 0xFFFFFFFFu : 0u;
        int jbA = (ImA >= D2) ? ((ImA - D2) >> 1) : 0;
        const v4u* xq = (const v4u*)(xrow + 32 * jbA + 8 * h);
        union { v4u q2[2]; unsigned u[8]; v16bf v; } B;
        B.q2[0] = xq[0];
        B.q2[1] = xq[2];
        #pragma unroll
        for (int zz = 0; zz < 8; ++zz) B.u[zz] &= valid;
        cacc[a] = __builtin_amdgcn_wmma_f32_16x16x32_bf16(
            false, A.v, false, B.v, (short)0, cacc[a], false, false);
      }
    }
  }

  // ---------------- epilogue: stage C tiles in LDS, coalesced b128 stores ---
  float* cw = &shc[wave][0];
  #pragma unroll
  for (int a = 0; a < 4; ++a)
    #pragma unroll
    for (int r = 0; r < 8; ++r)
      cw[a * 272 + i * 17 + 8 * h + r] = cacc[a][r];
  __syncthreads();

  {
    int pair = lane & 15;              // column = (batch, mm2)
    int part = lane >> 4;              // 8-row half
    int bb = pair & 3, mm2 = pair >> 2;
    bf16_t* yb = yT + ((size_t)bb * DCH + d) * LSEQ;
    #pragma unroll
    for (int a = 0; a < 4; ++a) {
      int t0 = 16 * (Ibase + 8 * a + 2 * mm2) + 8 * part;
      const float* cr = &shc[wave][a * 272 + pair * 17 + 8 * part];
      union { bf16_t hv[8]; v4u q; } OV;
      #pragma unroll
      for (int u2 = 0; u2 < 8; ++u2) OV.hv[u2] = (bf16_t)cr[u2];
      *(v4u*)(yb + t0) = OV.q;
    }
  }
}

// -------------------------------------------------------------------
// Kernel 5: y[b, t, d] = f32(yT[b, d, t]) * exp(-t/(L-1) * |delta_d|)
// -------------------------------------------------------------------
__global__ __launch_bounds__(256) void yfinal_kernel(
    const bf16_t* __restrict__ yT, float* __restrict__ y) {
  __shared__ float tile[32][33];
  int dT = blockIdx.x * 32;
  int tT = blockIdx.y * 32;
  int b  = blockIdx.z;
  int lx = threadIdx.x & 31;
  int ly = threadIdx.x >> 5;
  #pragma unroll
  for (int r = 0; r < 32; r += 8) {
    int d = dT + ly + r;
    int t = tT + lx;
    tile[ly + r][lx] = (float)yT[((size_t)b * DCH + d) * LSEQ + t];
  }
  __syncthreads();
  const float dmin = -9.210340371976184f;   // log(1e-2)/0.5
  const float dmax = -9.210340371976184f;
  #pragma unroll
  for (int r = 0; r < 32; r += 8) {
    int t = tT + ly + r;
    int d = dT + lx;
    float delta = dmin + (dmax - dmin) * ((float)d / (float)(DCH - 1));
    float dec = __expf(-((float)t / (float)(LSEQ - 1)) * fabsf(delta));
    y[((size_t)b * LSEQ + t) * DCH + d] = tile[lx][ly + r] * dec;
  }
}

// -------------------------------------------------------------------
// Launcher. Workspace layout (bytes):
//   H  : [0,            2 MB)   f32 [L, 64]
//   kT : [2 MB,        18 MB)   bf16 [D, L]
//   xT : [18 MB,       82 MB)   bf16 [B, D, L]
//   yT : [82 MB,      146 MB)   bf16 [B, D, L]
// -------------------------------------------------------------------
extern "C" void kernel_launch(void* const* d_in, const int* in_sizes, int n_in,
                              void* d_out, int out_size, void* d_ws, size_t ws_size,
                              hipStream_t stream) {
  const float* x  = (const float*)d_in[0];
  const float* pe = (const float*)d_in[1];
  const float* w1 = (const float*)d_in[2];
  const float* b1 = (const float*)d_in[3];
  const float* w2 = (const float*)d_in[4];
  const float* b2 = (const float*)d_in[5];
  const float* w3 = (const float*)d_in[6];
  const float* b3 = (const float*)d_in[7];
  float* y = (float*)d_out;

  char*   ws = (char*)d_ws;
  float*  H  = (float*)(ws);
  bf16_t* kT = (bf16_t*)(ws + (size_t)2 * 1024 * 1024);
  bf16_t* xT = (bf16_t*)(ws + (size_t)18 * 1024 * 1024);
  bf16_t* yT = (bf16_t*)(ws + (size_t)82 * 1024 * 1024);

  hfilt_kernel<<<dim3(LSEQ / 256), 256, 0, stream>>>(w1, b1, w2, b2, H);
  kproj_kernel<<<dim3(LSEQ / 256, DCH), 256, 0, stream>>>(H, w3, b3, kT);
  xtrans_kernel<<<dim3(DCH / 32, LSEQ / 32, NBATCH), 256, 0, stream>>>(x, pe, xT);
  hyena_conv_wmma_kernel<<<dim3(DCH * 8), 128, 0, stream>>>(kT, xT, yT);
  yfinal_kernel<<<dim3(DCH / 32, LSEQ / 32, NBATCH), 256, 0, stream>>>(yT, y);
}
